// Encoder_30374008717730
// MI455X (gfx1250) — matrix operand order repaired and checked by reading, not compiled, not run on hardware
//
#include <hip/hip_runtime.h>
#include <hip/hip_bf16.h>
#include <math.h>

// RAN encoder for MI455X (gfx1250, wave32, WMMA).
// D=512, L=512, B=64, 4 scans (2 layers x fw/bw).
// Bulk math: v_wmma_f32_16x16x32_bf16 (bf16 in, f32 accumulate).
// Cell state c and residual x kept in f32 for accuracy.

#define DD  512      // model dim
#define LL  512      // sequence length
#define BB  64       // batch
#define ND3 1536     // 3*D (gx width)
#define ND2 1024     // 2*D (gcarry width)
#define MM  (LL*BB)  // 32768 rows for the big GEMM

typedef __bf16 bf16;
typedef __attribute__((ext_vector_type(16))) __bf16 v16bf;
typedef __attribute__((ext_vector_type(8)))  __bf16 v8bf;
typedef __attribute__((ext_vector_type(8)))  float  v8f;

union V16U { v16bf v; v8bf h[2]; };

__device__ __forceinline__ v8f wmma_bf16(v16bf a, v16bf b, v8f c) {
  // (neg_a, A, neg_b, B, c_mod, C, reuse_a, reuse_b)
  return __builtin_amdgcn_wmma_f32_16x16x32_bf16(false, a, false, b, (short)0, c,
                                                 false, false);
}

// A fragment (16x32 bf16, MxK): lane holds row M = lane&15.
// lanes 0-15: K = kb+0..15
// lanes 16-31: K = kb+16..31
__device__ __forceinline__ v16bf load_a_frag(const bf16* __restrict__ A, int kb, int lane) {
  int row = lane & 15;
  int k0  = kb + (lane & 16);          // +16 for upper half-wave
  V16U u;
  const bf16* p = A + (size_t)row * DD + k0;
  u.h[0] = *(const v8bf*)(p);
  u.h[1] = *(const v8bf*)(p + 8);
  return u.v;
}

// B fragment (32x16 bf16, KxN) from a TRANSPOSED weight tile WT[N][K] (row-major,
// ld = 512): lane holds column N = lane&15; lanes 0-15 K=kb..kb+15, lanes 16-31
// K=kb+16..kb+31 -> one contiguous 32-byte load per lane.
__device__ __forceinline__ v16bf load_b_frag(const bf16* __restrict__ WT, int kb, int lane) {
  int col = lane & 15;
  int kh  = kb + (lane & 16);          // +16 for upper half-wave
  return *(const v16bf*)(WT + (size_t)col * DD + kh);
}

// ---------------------------------------------------------------------------
// Weight convert + transpose to bf16: WxT[s][n][k] = Wx[s][k][n]
// ---------------------------------------------------------------------------
__global__ __launch_bounds__(256) void conv_wx_kernel(const float* __restrict__ Wx,
                                                      bf16* __restrict__ WxT) {
  size_t i = (size_t)blockIdx.x * blockDim.x + threadIdx.x;
  if (i >= (size_t)4 * DD * ND3) return;
  int s   = (int)(i / ((size_t)DD * ND3));
  int rem = (int)(i % ((size_t)DD * ND3));
  int k = rem / ND3;
  int n = rem % ND3;
  WxT[((size_t)s * ND3 + n) * DD + k] = (bf16)Wx[i];
}

__global__ __launch_bounds__(256) void conv_wc_kernel(const float* __restrict__ Wc,
                                                      bf16* __restrict__ WcT) {
  size_t i = (size_t)blockIdx.x * blockDim.x + threadIdx.x;
  if (i >= (size_t)4 * DD * ND2) return;
  int s   = (int)(i / ((size_t)DD * ND2));
  int rem = (int)(i % ((size_t)DD * ND2));
  int k = rem / ND2;
  int n = rem % ND2;
  WcT[((size_t)s * ND2 + n) * DD + k] = (bf16)Wc[i];
}

// ---------------------------------------------------------------------------
// Embedding gather -> f32 sequence + bf16 copy for the GEMM
// ---------------------------------------------------------------------------
__global__ __launch_bounds__(256) void embed_kernel(const int* __restrict__ xs,
                                                    const float* __restrict__ emb,
                                                    float* __restrict__ xf,
                                                    bf16* __restrict__ xb) {
  size_t i = (size_t)blockIdx.x * blockDim.x + threadIdx.x;
  if (i >= (size_t)MM * DD) return;
  size_t row = i >> 9;          // / 512
  int d      = (int)(i & 511);
  int tok    = xs[row];
  float v    = emb[(size_t)tok * DD + d];
  xf[i] = v;
  xb[i] = (bf16)v;
}

// ---------------------------------------------------------------------------
// Zero initial cell state (f32 + bf16 copies, parity-0 buffers)
// ---------------------------------------------------------------------------
__global__ __launch_bounds__(256) void zero_c_kernel(float* __restrict__ cf,
                                                     bf16* __restrict__ cb) {
  int i = blockIdx.x * blockDim.x + threadIdx.x;
  if (i >= BB * DD) return;
  cf[i] = 0.0f;
  cb[i] = (bf16)0.0f;
}

// ---------------------------------------------------------------------------
// Big GEMM: gx[M x 1536] = X[M x 512](bf16) @ W[512 x 1536](as WT) + bias
// Block = 128 threads (4 waves); block tile 32(M) x 256(N); wave tile 32x64.
// Per K-chunk: 2 A frags + 4 B frags (12 x b128 loads) feed 8 WMMAs.
// ---------------------------------------------------------------------------
__global__ __launch_bounds__(128) void gx_gemm_kernel(const bf16* __restrict__ X,
                                                      const bf16* __restrict__ WT,
                                                      const float* __restrict__ bias,
                                                      float* __restrict__ gx) {
  const int lane = threadIdx.x & 31;
  const int wave = threadIdx.x >> 5;
  const int m0 = blockIdx.x * 32;
  const int n0 = blockIdx.y * 256 + wave * 64;

  v8f acc0[4] = {{}, {}, {}, {}};   // rows m0..m0+15   x 4 N-tiles
  v8f acc1[4] = {{}, {}, {}, {}};   // rows m0+16..+31  x 4 N-tiles
  const bf16* A0 = X + (size_t)m0 * DD;
  const bf16* A1 = A0 + (size_t)16 * DD;
  const bf16* Bp0 = WT + (size_t)n0 * DD;

#pragma unroll 2
  for (int kb = 0; kb < DD; kb += 32) {
    v16bf a0 = load_a_frag(A0, kb, lane);
    v16bf a1 = load_a_frag(A1, kb, lane);
#pragma unroll
    for (int j = 0; j < 4; ++j) {
      v16bf bj = load_b_frag(Bp0 + (size_t)(16 * j) * DD, kb, lane);
      acc0[j] = wmma_bf16(a0, bj, acc0[j]);
      acc1[j] = wmma_bf16(a1, bj, acc1[j]);
    }
  }

  const int col    = lane & 15;
  const int rowoff = (lane & 16) >> 1;   // +8 for upper half-wave
#pragma unroll
  for (int j = 0; j < 4; ++j) {
    const int n = n0 + 16 * j + col;
    const float bv = bias[n];
#pragma unroll
    for (int v = 0; v < 8; ++v) {
      int mA = m0 + rowoff + v;
      gx[(size_t)mA * ND3 + n]        = acc0[j][v] + bv;
      gx[(size_t)(mA + 16) * ND3 + n] = acc1[j][v] + bv;
    }
  }
}

// ---------------------------------------------------------------------------
// One RAN recurrence step at time t.
// 128 waves total (32 blocks x 4 waves). Wave (bm,dn) computes a 16x16 tile of
// gcarry_i (cols dn*16) and gcarry_f (cols D + dn*16) with K=512, then the
// gates / cell update / residual output for that tile.
// ---------------------------------------------------------------------------
__global__ __launch_bounds__(128) void ran_step_kernel(
    const bf16*  __restrict__ cbf_in,   // [64 x 512] bf16 cell state (read)
    const float* __restrict__ cf_in,    // [64 x 512] f32 cell state (read)
    bf16*        __restrict__ cbf_out,  // [64 x 512] (write)
    float*       __restrict__ cf_out,   // [64 x 512] (write)
    const bf16*  __restrict__ WcT,      // [1024 x 512] transposed Wc for scan
    const float* __restrict__ gx,       // [L x 64 x 1536] precomputed x-gates
    const float* __restrict__ xin,      // [L x 64 x 512] f32 residual input
    float*       __restrict__ hout,     // [L x 64 x 512] f32 output
    bf16*        __restrict__ hbf,      // [L x 64 x 512] bf16 output (next GEMM)
    int t) {
  const int lane = threadIdx.x & 31;
  const int wid  = blockIdx.x * 4 + (threadIdx.x >> 5);  // 0..127
  const int bm   = wid >> 5;                             // 0..3  (batch tile)
  const int dn   = wid & 31;                             // 0..31 (d tile)

  v8f acc_i = {}, acc_f = {};
  const bf16* Arow = cbf_in + (size_t)(bm * 16) * DD;
  const bf16* Bi   = WcT + (size_t)(dn * 16) * DD;
  const bf16* Bf   = WcT + (size_t)(DD + dn * 16) * DD;

#pragma unroll 4
  for (int kb = 0; kb < DD; kb += 32) {
    v16bf a = load_a_frag(Arow, kb, lane);
    acc_i = wmma_bf16(a, load_b_frag(Bi, kb, lane), acc_i);
    acc_f = wmma_bf16(a, load_b_frag(Bf, kb, lane), acc_f);
  }

  const int col    = lane & 15;
  const int rowoff = (lane & 16) >> 1;
  const size_t tb  = (size_t)t * BB;
#pragma unroll
  for (int v = 0; v < 8; ++v) {
    int bi = bm * 16 + rowoff + v;
    int di = dn * 16 + col;
    const float* gp = gx + (tb + (size_t)bi) * ND3;
    float gi = gp[di];
    float gf = gp[DD + di];
    float gc = gp[2 * DD + di];
    float co = cf_in[(size_t)bi * DD + di];
    float ig = 1.0f / (1.0f + __expf(-(gi + acc_i[v])));
    float fg = 1.0f / (1.0f + __expf(-(gf + acc_f[v])));
    float cn = ig * gc + fg * co;
    float hv = tanhf(cn) + xin[(tb + (size_t)bi) * DD + di];
    cf_out[(size_t)bi * DD + di]  = cn;
    cbf_out[(size_t)bi * DD + di] = (bf16)cn;
    hout[(tb + (size_t)bi) * DD + di] = hv;
    hbf[(tb + (size_t)bi) * DD + di]  = (bf16)hv;
  }
}

// ---------------------------------------------------------------------------
// Host orchestration
// ---------------------------------------------------------------------------
extern "C" void kernel_launch(void* const* d_in, const int* in_sizes, int n_in,
                              void* d_out, int out_size, void* d_ws, size_t ws_size,
                              hipStream_t stream) {
  (void)in_sizes; (void)n_in; (void)out_size; (void)ws_size;
  const int*   xs   = (const int*)d_in[0];
  const float* emb  = (const float*)d_in[1];
  const float* Wx   = (const float*)d_in[2];
  const float* Wc   = (const float*)d_in[3];
  const float* bias = (const float*)d_in[4];

  char* ws = (char*)d_ws;
  size_t off = 0;
  auto alloc = [&](size_t bytes) -> void* {
    off = (off + 255) & ~(size_t)255;
    void* p = ws + off;
    off += bytes;
    return p;
  };

  float* xf[2];   // f32 sequence ping-pong
  bf16*  xb[2];   // bf16 sequence ping-pong
  xf[0] = (float*)alloc((size_t)MM * DD * 4);
  xf[1] = (float*)alloc((size_t)MM * DD * 4);
  xb[0] = (bf16*)alloc((size_t)MM * DD * 2);
  xb[1] = (bf16*)alloc((size_t)MM * DD * 2);
  float* gx  = (float*)alloc((size_t)MM * ND3 * 4);
  bf16*  WxT = (bf16*)alloc((size_t)4 * ND3 * DD * 2);
  bf16*  WcT = (bf16*)alloc((size_t)4 * ND2 * DD * 2);
  float* cf[2];
  bf16*  cb[2];
  cf[0] = (float*)alloc((size_t)BB * DD * 4);
  cf[1] = (float*)alloc((size_t)BB * DD * 4);
  cb[0] = (bf16*)alloc((size_t)BB * DD * 2);
  cb[1] = (bf16*)alloc((size_t)BB * DD * 2);

  // Weight convert + transpose (once per call).
  {
    size_t nwx = (size_t)4 * DD * ND3;
    size_t nwc = (size_t)4 * DD * ND2;
    conv_wx_kernel<<<(unsigned)((nwx + 255) / 256), 256, 0, stream>>>(Wx, WxT);
    conv_wc_kernel<<<(unsigned)((nwc + 255) / 256), 256, 0, stream>>>(Wc, WcT);
  }
  // Embedding gather.
  {
    size_t ntot = (size_t)MM * DD;
    embed_kernel<<<(unsigned)((ntot + 255) / 256), 256, 0, stream>>>(xs, emb, xf[0], xb[0]);
  }
  // Zero initial cell state (parity 0).
  zero_c_kernel<<<(BB * DD + 255) / 256, 256, 0, stream>>>(cf[0], cb[0]);

  int par = 0;  // cell-state parity (carries across scans)
  int cur = 0;  // sequence buffer parity
  for (int scan = 0; scan < 4; ++scan) {
    const bf16*  WxT_s  = WxT + (size_t)scan * ND3 * DD;
    const bf16*  WcT_s  = WcT + (size_t)scan * ND2 * DD;
    const float* bias_s = bias + scan * ND3;

    // gx = X @ Wx + b for the whole sequence (parallel WMMA GEMM).
    dim3 ggrid(MM / 32, ND3 / 256);
    gx_gemm_kernel<<<ggrid, 128, 0, stream>>>(xb[cur], WxT_s, bias_s, gx);

    const bool reverse = (scan & 1) != 0;
    const bool last    = (scan == 3);
    float* hf = last ? (float*)d_out : xf[cur ^ 1];
    bf16*  hb = xb[cur ^ 1];

    for (int s = 0; s < LL; ++s) {
      int t = reverse ? (LL - 1 - s) : s;
      ran_step_kernel<<<32, 128, 0, stream>>>(cb[par], cf[par],
                                              cb[par ^ 1], cf[par ^ 1],
                                              WcT_s, gx, xf[cur], hf, hb, t);
      par ^= 1;
    }
    cur ^= 1;
  }
}
